// PerformerAttention_23759759081761
// MI455X (gfx1250) — matrix operand order repaired and checked by reading, not compiled, not run on hardware
//
#include <hip/hip_runtime.h>
#include <math.h>

typedef __attribute__((ext_vector_type(16))) _Float16 v16h;
typedef __attribute__((ext_vector_type(8)))  _Float16 v8h;
typedef __attribute__((ext_vector_type(8)))  float    v8f;

#define B_    2
#define H_    16
#define L_    4096
#define D_    64
#define M_    266      // real feature count
#define MP_   272      // padded to 17*16
#define MP2_  288      // padded to 9*32 (K chunks for second GEMM)
#define BH_   32
#define ROWS_ 64       // L rows per block
#define HD_   1024     // H*D

#define DN_     0.35355339059327379f  // 64^-0.25 (folded into Ppad16)
#define HDN2_   0.0625f               // 0.5 * DN^2 = 0.5 * 2^-3 (exact)
#define RATIO_  0.06131395f           // 266^-0.5
#define EPS_    1e-4f

// ---------- ordered-uint encoding for float atomicMax ----------
static __device__ inline unsigned enc_ord(float f) {
  unsigned u = __float_as_uint(f);
  return (u & 0x80000000u) ? ~u : (u | 0x80000000u);
}
static __device__ inline float dec_ord(unsigned e) {
  return (e & 0x80000000u) ? __uint_as_float(e & 0x7FFFFFFFu)
                           : __uint_as_float(~e);
}

// ---------- WMMA wrapper ----------
static __device__ inline v8f wmma32(v16h a, v16h b, v8f c) {
  return __builtin_amdgcn_wmma_f32_16x16x32_f16(
      false, a, false, b, (short)0, c, false, false);
}

// ---------- async global->LDS staging (gfx1250 ASYNC path) ----------
// Copies 64x64 f32 tile (row-major, contiguous in global) into LDS with
// row stride 68 floats. 8 x b128 per thread, tracked by ASYNCcnt.
static __device__ inline void async_stage_tile(const float* gsrc, float* lds,
                                               int tid) {
  unsigned lds0 = (unsigned)(uintptr_t)lds;   // low 32 bits = LDS address
#pragma unroll
  for (int j = 0; j < 8; ++j) {
    int f0 = (j * 128 + tid) << 2;            // 4 floats per op
    int r = f0 >> 6, d = f0 & 63;
    unsigned loff = lds0 + (unsigned)((r * 68 + d) * 4);
    unsigned long long ga = (unsigned long long)(uintptr_t)(gsrc + f0);
    asm volatile("global_load_async_to_lds_b128 %0, %1, off"
                 :: "v"(loff), "v"(ga) : "memory");
  }
}
static __device__ inline void async_wait_all() {
  asm volatile("s_wait_asynccnt 0" ::: "memory");
}

// ---------- fragment loaders (wave32 layouts, ISA 7.12.2) ----------
// A (16xK) from row-major f32 src[row][k] with conversion.
static __device__ inline v16h load_a_frag(const float* src, int ld,
                                          int row0, int k0, int lane) {
  int h2  = (lane >> 4) & 1;
  int row = row0 + (lane & 15);
  const float* p = src + row * ld + k0 + 8 * h2;
  v16h a;
#pragma unroll
  for (int j = 0; j < 8; ++j) a[j] = (_Float16)p[j];
#pragma unroll
  for (int j = 0; j < 8; ++j) a[8 + j] = (_Float16)p[16 + j];
  return a;
}

// A (16xK) from row-major f16 src[row][k]: two contiguous 8-half chunks.
static __device__ inline v16h load_a16(const _Float16* src, int ld,
                                       int row0, int k0, int lane) {
  int h2  = (lane >> 4) & 1;
  int row = row0 + (lane & 15);
  const _Float16* p = src + row * ld + k0 + 8 * h2;
  v8h lo = *(const v8h*)p;
  v8h hi = *(const v8h*)(p + 16);
  return __builtin_shufflevector(lo, hi, 0, 1, 2, 3, 4, 5, 6, 7,
                                 8, 9, 10, 11, 12, 13, 14, 15);
}

// B (Kx16) where B[k][n] = src[n][k], src row-major f16 [n][k]:
// two contiguous 8-half chunks (k0+8*h2 and k0+16+8*h2).
static __device__ inline v16h load_b16T(const _Float16* src, int ld,
                                        int n0, int k0, int lane) {
  int h2 = (lane >> 4) & 1;
  int n  = n0 + (lane & 15);
  const _Float16* p = src + n * ld + k0 + 8 * h2;
  v8h lo = *(const v8h*)p;
  v8h hi = *(const v8h*)(p + 16);
  return __builtin_shufflevector(lo, hi, 0, 1, 2, 3, 4, 5, 6, 7,
                                 8, 9, 10, 11, 12, 13, 14, 15);
}

// ---------------------------------------------------------------
// Kernel 0: init workspace (k-max scalar, k_sum, context, f16 DN*P)
// ---------------------------------------------------------------
__global__ void perf_prep(const float* __restrict__ proj,
                          _Float16* __restrict__ Ppad16,
                          float* __restrict__ ksum, float* __restrict__ ctx,
                          unsigned* __restrict__ kmax) {
  int tid = blockIdx.x * blockDim.x + threadIdx.x;
  int stride = gridDim.x * blockDim.x;
  if (tid == 0) *kmax = 0x007FFFFFu;  // enc(-inf)
  for (int i = tid; i < MP2_ * D_; i += stride) {
    int m = i >> 6, d = i & 63;
    Ppad16[i] = (m < M_) ? (_Float16)(DN_ * proj[m * D_ + d]) : (_Float16)0.f;
  }
  for (int i = tid; i < BH_ * MP_; i += stride) ksum[i] = 0.f;
  for (int i = tid; i < BH_ * MP_ * D_; i += stride) ctx[i] = 0.f;
}

// ---------------------------------------------------------------
// Kernel 1: global max of k's data_dash
// ---------------------------------------------------------------
__global__ __launch_bounds__(128)
void perf_kmax(const float* __restrict__ k, const _Float16* __restrict__ Ppad16,
               unsigned* __restrict__ kmax) {
  extern __shared__ char smraw[];
  float* sk  = (float*)smraw;              // [64][68] raw k tile
  float* red = (float*)(smraw + 17408);    // [128]
  int tid = threadIdx.x, lane = tid & 31, wv = tid >> 5;
  int lt = blockIdx.x, bh = blockIdx.y;
  long base = ((long)bh * L_ + (long)lt * ROWS_) * D_;

  async_stage_tile(k + base, sk, tid);
  async_wait_all();
  __syncthreads();

  float tmax = -3.0e38f;
  int row0 = wv * 16, col = lane & 15;
  v16h a0 = load_a_frag(sk, 68, row0, 0, lane);
  v16h a1 = load_a_frag(sk, 68, row0, 32, lane);
  for (int mt = 0; mt < 17; ++mt) {
    v8f c = {};
    c = wmma32(a0, load_b16T(Ppad16, D_, mt * 16, 0, lane), c);
    c = wmma32(a1, load_b16T(Ppad16, D_, mt * 16, 32, lane), c);
    if (mt * 16 + col < M_) {
#pragma unroll
      for (int i2 = 0; i2 < 8; ++i2) tmax = fmaxf(tmax, c[i2]);
    }
  }
  red[tid] = tmax;
  __syncthreads();
  for (int s = 64; s > 0; s >>= 1) {
    if (tid < s) red[tid] = fmaxf(red[tid], red[tid + s]);
    __syncthreads();
  }
  if (tid == 0) atomicMax(kmax, enc_ord(red[0]));
}

// ---------------------------------------------------------------
// Kernel 2: phi(k), k_sum, context = phi(k)^T @ v  (atomic partials)
// ---------------------------------------------------------------
__global__ __launch_bounds__(128)
void perf_kcontext(const float* __restrict__ k, const float* __restrict__ v,
                   const unsigned char* __restrict__ mask,
                   const _Float16* __restrict__ Ppad16,
                   const unsigned* __restrict__ kmaxp,
                   float* __restrict__ ksum, float* __restrict__ ctx) {
  extern __shared__ char smraw[];
  float*     sk    = (float*)smraw;                 // [64][68]   raw k
  _Float16*  vT16  = (_Float16*)(smraw + 17408);    // [64d][72l] masked v^T
  _Float16*  kpT16 = (_Float16*)(smraw + 26624);    // [272m][72l] phi(k)^T
  float*     diag  = (float*)(smraw + 65792);       // [64]
  float*     red   = (float*)(smraw + 66048);       // [128]
  int tid = threadIdx.x, lane = tid & 31, wv = tid >> 5;
  int lt = blockIdx.x, bh = blockIdx.y;
  int b = bh >> 4;
  long base = ((long)bh * L_ + (long)lt * ROWS_) * D_;
  const unsigned char* mrow = mask + (long)b * L_ + (long)lt * ROWS_;

  async_stage_tile(k + base, sk, tid);          // overlap with v staging
  for (int i = tid; i < ROWS_ * D_; i += 128) { // v masked + transposed
    int r = i >> 6, d = i & 63;
    float val = mrow[r] ? v[base + i] : 0.f;
    vT16[d * 72 + r] = (_Float16)val;
  }
  async_wait_all();
  __syncthreads();
  {
    int r = tid >> 1, p = tid & 1;
    const float* pr = sk + r * 68 + p * 32;
    float s = 0.f;
#pragma unroll
    for (int j = 0; j < 32; ++j) s += pr[j] * pr[j];
    red[tid] = s;
  }
  __syncthreads();
  if ((tid & 1) == 0) diag[tid >> 1] = HDN2_ * (red[tid] + red[tid + 1]);
  __syncthreads();

  float kmx = dec_ord(*kmaxp);
  int row0 = wv * 16, h2 = (lane >> 4) & 1, col = lane & 15;
  v16h a0 = load_a_frag(sk, 68, row0, 0, lane);
  v16h a1 = load_a_frag(sk, 68, row0, 32, lane);
  for (int mt = 0; mt < 17; ++mt) {
    v8f c = {};
    c = wmma32(a0, load_b16T(Ppad16, D_, mt * 16, 0, lane), c);
    c = wmma32(a1, load_b16T(Ppad16, D_, mt * 16, 32, lane), c);
    int m = mt * 16 + col;
#pragma unroll
    for (int i2 = 0; i2 < 8; ++i2) {
      int r = row0 + i2 + 8 * h2;
      float val = (m < M_) ? RATIO_ * (__expf(c[i2] - diag[r] - kmx) + EPS_)
                           : 0.f;
      kpT16[m * 72 + r] = (_Float16)val;
    }
  }
  __syncthreads();

  // k_sum partials
  for (int m = tid; m < MP_; m += 128) {
    float s = 0.f;
#pragma unroll 8
    for (int r = 0; r < ROWS_; ++r) s += (float)kpT16[m * 72 + r];
    atomicAdd(&ksum[bh * MP_ + m], s);
  }

  // context partials: [272 x 64] += phi(k)^T(16m x 64l) x v(64l x 16d)
  float* cg = ctx + (long)bh * MP_ * D_;
  for (int idx = wv; idx < 68; idx += 4) {   // 17 m-tiles x 4 d-tiles
    int mt = idx >> 2, dt = idx & 3;
    v8f c = {};
    c = wmma32(load_a16(kpT16, 72, mt * 16, 0, lane),
               load_b16T(vT16, 72, dt * 16, 0, lane), c);
    c = wmma32(load_a16(kpT16, 72, mt * 16, 32, lane),
               load_b16T(vT16, 72, dt * 16, 32, lane), c);
#pragma unroll
    for (int i2 = 0; i2 < 8; ++i2) {
      int m = mt * 16 + i2 + 8 * h2;
      atomicAdd(&cg[m * D_ + dt * 16 + col], c[i2]);
    }
  }
}

// ---------------------------------------------------------------
// Kernel 3: fused q path: project, row-max, phi(q), d_inv,
//           out = (phi(q) @ context) * d_inv, written as [B, L, H*D]
// ---------------------------------------------------------------
__global__ __launch_bounds__(128)
void perf_qout(const float* __restrict__ q,
               const _Float16* __restrict__ Ppad16,
               const float* __restrict__ ksumg, const float* __restrict__ ctxg,
               float* __restrict__ out) {
  extern __shared__ char smraw[];
  float*     sq     = (float*)smraw;                 // [64][68]    raw q
  float*     qp     = (float*)(smraw + 17408);       // [64][296]   dd (f32)
  _Float16*  qp16   = (_Float16*)(smraw + 93184);    // [64][296]   phi(q) f16
  _Float16*  ctxT16 = (_Float16*)(smraw + 131072);   // [64d][296m] context^T
  float*     ks     = (float*)(smraw + 168960);      // [288]
  float*     diag   = (float*)(smraw + 170112);      // [64]
  float*     rmax   = (float*)(smraw + 170368);      // [64]
  float*     dinv   = (float*)(smraw + 170624);      // [64]
  float*     red    = (float*)(smraw + 170880);      // [128]
  int tid = threadIdx.x, lane = tid & 31, wv = tid >> 5;
  int lt = blockIdx.x, bh = blockIdx.y;
  int b = bh >> 4, h = bh & 15;
  long base = ((long)bh * L_ + (long)lt * ROWS_) * D_;

  async_stage_tile(q + base, sq, tid);          // overlap with ctx staging
  for (int i = tid; i < MP2_ * D_; i += 128) {  // context transposed -> f16
    int m = i >> 6, d = i & 63;
    float val = (m < MP_) ? ctxg[(long)bh * MP_ * D_ + i] : 0.f;
    ctxT16[d * 296 + m] = (_Float16)val;
  }
  for (int i = tid; i < MP2_; i += 128)
    ks[i] = (i < MP_) ? ksumg[bh * MP_ + i] : 0.f;
  async_wait_all();
  __syncthreads();
  {
    int r = tid >> 1, p = tid & 1;
    const float* pr = sq + r * 68 + p * 32;
    float s = 0.f;
#pragma unroll
    for (int j = 0; j < 32; ++j) s += pr[j] * pr[j];
    red[tid] = s;
  }
  __syncthreads();
  if ((tid & 1) == 0) diag[tid >> 1] = HDN2_ * (red[tid] + red[tid + 1]);

  // GEMM1: dd = q @ (DN*P)^T  (raw f32 into qp)
  int row0 = wv * 16, h2 = (lane >> 4) & 1, col = lane & 15;
  v16h a0 = load_a_frag(sq, 68, row0, 0, lane);
  v16h a1 = load_a_frag(sq, 68, row0, 32, lane);
  for (int mt = 0; mt < 17; ++mt) {
    v8f c = {};
    c = wmma32(a0, load_b16T(Ppad16, D_, mt * 16, 0, lane), c);
    c = wmma32(a1, load_b16T(Ppad16, D_, mt * 16, 32, lane), c);
#pragma unroll
    for (int i2 = 0; i2 < 8; ++i2)
      qp[(row0 + i2 + 8 * h2) * 296 + mt * 16 + col] = c[i2];
  }
  __syncthreads();

  // per-row max over real features
  {
    int r = tid >> 1, p = tid & 1;
    const float* pr = qp + r * 296 + p * 136;
    int lim = p ? (M_ - 136) : 136;
    float mx = -3.0e38f;
    for (int j = 0; j < lim; ++j) mx = fmaxf(mx, pr[j]);
    red[tid] = mx;
  }
  __syncthreads();
  if ((tid & 1) == 0) rmax[tid >> 1] = fmaxf(red[tid], red[tid + 1]);
  __syncthreads();

  // phi(q) -> qp16, plus d = phi(q) . k_sum
  {
    int r = tid >> 1, p = tid & 1;
    float dg = diag[r], st = rmax[r], dacc = 0.f;
    const float* pr = qp + r * 296 + p * 136;
    const float* kk = ks + p * 136;
    _Float16* po = qp16 + r * 296 + p * 136;
    for (int j = 0; j < 136; ++j) {
      int m = p * 136 + j;
      float val = (m < M_) ? RATIO_ * (__expf(pr[j] - dg - st) + EPS_) : 0.f;
      po[j] = (_Float16)val;
      dacc += val * kk[j];
    }
    red[tid] = dacc;
  }
  for (int i = tid; i < ROWS_ * 16; i += 128)     // zero K-pad cols 272..287
    qp16[(i >> 4) * 296 + MP_ + (i & 15)] = (_Float16)0.f;
  __syncthreads();
  if ((tid & 1) == 0) dinv[tid >> 1] = 1.f / (red[tid] + red[tid + 1]);
  __syncthreads();

  // GEMM2: out = phi(q)[64 x 288] @ context[288 x 64]; kc-outer, 4 C tiles
  v8f cc[4] = {{}, {}, {}, {}};
#pragma unroll
  for (int kc = 0; kc < 9; ++kc) {
    v16h a = load_a16(qp16, 296, row0, kc * 32, lane);
#pragma unroll
    for (int dt = 0; dt < 4; ++dt)
      cc[dt] = wmma32(a, load_b16T(ctxT16, 296, dt * 16, kc * 32, lane), cc[dt]);
  }
  float* og = out + (long)b * L_ * HD_ + (long)(lt * ROWS_) * HD_ + h * D_;
#pragma unroll
  for (int dt = 0; dt < 4; ++dt) {
#pragma unroll
    for (int i2 = 0; i2 < 8; ++i2) {
      int r = row0 + i2 + 8 * h2;
      og[(long)r * HD_ + dt * 16 + col] = cc[dt][i2] * dinv[r];
    }
  }
}

// ---------------------------------------------------------------
extern "C" void kernel_launch(void* const* d_in, const int* in_sizes, int n_in,
                              void* d_out, int out_size, void* d_ws, size_t ws_size,
                              hipStream_t stream) {
  const float* q = (const float*)d_in[0];
  const float* k = (const float*)d_in[1];
  const float* v = (const float*)d_in[2];
  const unsigned char* mask = (const unsigned char*)d_in[3];  // bool [B, L]
  const float* proj = (const float*)d_in[4];                  // [266, 64]
  float* out = (float*)d_out;

  char* ws = (char*)d_ws;
  unsigned*  kmax   = (unsigned*)ws;                 // 4 B (pad to 256)
  float*     ksum   = (float*)(ws + 256);            // 32*272 f32   (34816 B)
  _Float16*  Ppad16 = (_Float16*)(ws + 35072);       // 288*64 f16   (36864 B)
  float*     ctx    = (float*)(ws + 71936);          // 32*272*64 f32

  size_t sm_k1 = 17920;    // sk(17408) + red(512)
  size_t sm_k2 = 66560;    // sk + vT16 + kpT16 + diag + red
  size_t sm_q  = 171392;   // sq + qp + qp16 + ctxT16 + ks + diag/rmax/dinv/red

  perf_prep<<<dim3(512), dim3(256), 0, stream>>>(proj, Ppad16, ksum, ctx, kmax);
  perf_kmax<<<dim3(L_ / ROWS_, BH_), dim3(128), sm_k1, stream>>>(k, Ppad16, kmax);
  perf_kcontext<<<dim3(L_ / ROWS_, BH_), dim3(128), sm_k2, stream>>>(
      k, v, mask, Ppad16, kmax, ksum, ctx);
  perf_qout<<<dim3(L_ / ROWS_, BH_), dim3(128), sm_q, stream>>>(
      q, Ppad16, ksum, ctx, out);
}
